// ConvFlow_55336358643237
// MI455X (gfx1250) — compile-verified, matrix operand order checked by reading, not run on hardware
//
#include <hip/hip_runtime.h>

#define BB   32
#define TT   4096
#define NTOT (BB * TT)     // 131072 positions
#define HH   256
#define NBINS 10
#define TAILB 5.0f

typedef __attribute__((ext_vector_type(16))) __bf16 v16bf;
typedef __attribute__((ext_vector_type(8)))  float  v8f;
typedef __attribute__((ext_vector_type(4)))  int    v4i;
typedef __attribute__((address_space(1))) v4i* gp4;   // global int4*
typedef __attribute__((address_space(3))) v4i* lp4;   // LDS int4*

#if defined(__has_builtin)
#if __has_builtin(__builtin_amdgcn_global_load_async_to_lds_b128) && \
    __has_builtin(__builtin_amdgcn_global_store_async_from_lds_b128)
#define HAVE_ASYNC 1
#endif
#endif
#ifndef HAVE_ASYNC
#define HAVE_ASYNC 0
#endif

// 16-byte global -> LDS copy (async DMA on CDNA5, VGPR bounce otherwise)
__device__ __forceinline__ void g2l16(const void* g, void* l) {
#if HAVE_ASYNC
  __builtin_amdgcn_global_load_async_to_lds_b128((gp4)g, (lp4)l, 0, 0);
#else
  *(uint4*)l = *(const uint4*)g;
#endif
}
// 16-byte LDS -> global copy
__device__ __forceinline__ void l2g16(void* g, const void* l) {
#if HAVE_ASYNC
  __builtin_amdgcn_global_store_async_from_lds_b128((gp4)g, (lp4)l, 0, 0);
#else
  *(uint4*)g = *(const uint4*)l;
#endif
}
__device__ __forceinline__ void wait_async0() {
#if HAVE_ASYNC
#if __has_builtin(__builtin_amdgcn_s_wait_asynccnt)
  __builtin_amdgcn_s_wait_asynccnt(0);
#else
  asm volatile("s_wait_asynccnt 0x0" ::: "memory");
#endif
#endif
}

__device__ __forceinline__ unsigned short f2bf(float f) {
  unsigned int u = __float_as_uint(f);
  u += 0x7FFFu + ((u >> 16) & 1u);   // round-to-nearest-even
  return (unsigned short)(u >> 16);
}
__device__ __forceinline__ float gelu_exact(float v) {
  return 0.5f * v * (1.0f + erff(v * 0.70710678118654752f));
}
__device__ __forceinline__ float softplusf(float x) {
  return (x > 20.0f) ? x : log1pf(__expf(x));
}

// h[n,o] = pre_w[o,0]*x0a + pre_w[o,1]*x0b + pre_b[o]   (channels-last)
__global__ __launch_bounds__(256) void k_pre(const float* __restrict__ x,
    const float* __restrict__ pre_w, const float* __restrict__ pre_b,
    float* __restrict__ h) {
  const int n = blockIdx.x;
  const int o = threadIdx.x;
  const int b = n >> 12, t = n & (TT - 1);
  const float x0a = x[((size_t)(b * 4 + 0)) * TT + t];
  const float x0b = x[((size_t)(b * 4 + 1)) * TT + t];
  h[(size_t)n * HH + o] = pre_w[o * 2] * x0a + pre_w[o * 2 + 1] * x0b + pre_b[o];
}

// depthwise conv (K=3, dilation dil) + layernorm over channels + exact GELU -> bf16
__global__ __launch_bounds__(256) void k_dwln(const float* __restrict__ h,
    const float* __restrict__ xmask,
    const float* __restrict__ dw, const float* __restrict__ db,
    const float* __restrict__ g1, const float* __restrict__ b1,
    unsigned short* __restrict__ xb, int dil) {
  __shared__ float s1[256], s2[256];
  const int n = blockIdx.x, c = threadIdx.x;
  const int t = n & (TT - 1);
  float acc = db[c];
  const float w0 = dw[c * 3], w1 = dw[c * 3 + 1], w2 = dw[c * 3 + 2];
  if (t >= dil)     { const int nn = n - dil; acc += w0 * h[(size_t)nn * HH + c] * xmask[nn]; }
  acc += w1 * h[(size_t)n * HH + c] * xmask[n];
  if (t + dil < TT) { const int nn = n + dil; acc += w2 * h[(size_t)nn * HH + c] * xmask[nn]; }
  s1[c] = acc; s2[c] = acc * acc;
  __syncthreads();
  for (int s = 128; s > 0; s >>= 1) {
    if (c < s) { s1[c] += s1[c + s]; s2[c] += s2[c + s]; }
    __syncthreads();
  }
  const float mean = s1[0] * (1.0f / HH);
  const float var  = s2[0] * (1.0f / HH) - mean * mean;
  const float yn = (acc - mean) * rsqrtf(var + 1e-5f) * g1[c] + b1[c];
  xb[(size_t)n * HH + c] = f2bf(gelu_exact(yn));
}

// f32 -> bf16 convert (weights)
__global__ __launch_bounds__(256) void k_cvt(const float* __restrict__ src,
    unsigned short* __restrict__ dst, int nelem) {
  const int i = blockIdx.x * 256 + threadIdx.x;
  if (i < nelem) dst[i] = f2bf(src[i]);
}

// proj weights -> bf16 padded to 64 rows
__global__ __launch_bounds__(256) void k_projw(const float* __restrict__ pw,
    unsigned short* __restrict__ pwb) {
  const int o = blockIdx.x, c = threadIdx.x;
  pwb[o * HH + c] = (o < 58) ? f2bf(pw[o * HH + c]) : (unsigned short)0;
}

// 256x256 x 256xN GEMM via v_wmma_f32_16x16x32_bf16.
// Block = 8 waves. X-tile (32 n-rows x 256 K) async-staged to LDS once per block;
// all waves feed B-fragments from LDS. D tiles staged in LDS, then async-stored
// to (n,c) layout. Bias fused into C init.
__global__ __launch_bounds__(256) void k_gemm(const unsigned short* __restrict__ wb,
    const unsigned short* __restrict__ xb, const float* __restrict__ bias,
    float* __restrict__ yt) {
  __shared__ unsigned short xs[32][HH];   // 16 KB X tile
  __shared__ float tile[32][128];         // 16 KB output staging
  const int tid = threadIdx.x, wave = tid >> 5, lane = tid & 31;
  const int lhi = lane >> 4, llo = lane & 15;
  const int o0b = blockIdx.y * 128;
  const int o0  = o0b + wave * 16;
  const int n0  = blockIdx.x * 32;

  // stage X tile: 1024 x 16B chunks, 4 per thread
#pragma unroll
  for (int j = 0; j < 4; ++j) {
    const int q  = tid + j * 256;
    const int rr = q >> 5;            // row in tile
    const int cc = (q & 31) * 8;      // halves within row
    g2l16(xb + (size_t)(n0 + rr) * HH + cc, &xs[rr][cc]);
  }
  wait_async0();
  __syncthreads();

  v8f c0, c1;
#pragma unroll
  for (int r = 0; r < 8; ++r) {
    const float bv = bias[o0 + r + 8 * lhi];
    c0[r] = bv; c1[r] = bv;
  }
  const int row = o0 + llo;
#pragma unroll
  for (int kk = 0; kk < 8; ++kk) {
    const int koff = kk * 32 + lhi * 16;
    v16bf a  = *(const v16bf*)(wb + row * HH + koff);
    v16bf b0 = *(const v16bf*)(&xs[llo][koff]);
    v16bf b1 = *(const v16bf*)(&xs[16 + llo][koff]);
    c0 = __builtin_amdgcn_wmma_f32_16x16x32_bf16(false, a, false, b0, (short)0, c0, false, false);
    c1 = __builtin_amdgcn_wmma_f32_16x16x32_bf16(false, a, false, b1, (short)0, c1, false, false);
  }
  float* d0 = &tile[llo][wave * 16 + 8 * lhi];
  *(float4*)d0       = make_float4(c0[0], c0[1], c0[2], c0[3]);
  *(float4*)(d0 + 4) = make_float4(c0[4], c0[5], c0[6], c0[7]);
  float* d1 = &tile[16 + llo][wave * 16 + 8 * lhi];
  *(float4*)d1       = make_float4(c1[0], c1[1], c1[2], c1[3]);
  *(float4*)(d1 + 4) = make_float4(c1[4], c1[5], c1[6], c1[7]);
  __syncthreads();
  // stream output tile LDS -> global (coalesced, 16B chunks)
#pragma unroll
  for (int j = 0; j < 4; ++j) {
    const int q   = tid + j * 256;
    const int rr  = q >> 5;           // 32 chunks per 128-float row
    const int cc4 = (q & 31) * 4;
    l2g16(yt + (size_t)(n0 + rr) * HH + o0b + cc4, &tile[rr][cc4]);
  }
}

// layernorm + GELU + residual accumulate into h
__global__ __launch_bounds__(256) void k_ln2(const float* __restrict__ yt,
    const float* __restrict__ g2, const float* __restrict__ b2,
    float* __restrict__ h) {
  __shared__ float s1[256], s2[256];
  const int n = blockIdx.x, c = threadIdx.x;
  const float y = yt[(size_t)n * HH + c];
  s1[c] = y; s2[c] = y * y;
  __syncthreads();
  for (int s = 128; s > 0; s >>= 1) {
    if (c < s) { s1[c] += s1[c + s]; s2[c] += s2[c + s]; }
    __syncthreads();
  }
  const float mean = s1[0] * (1.0f / HH);
  const float var  = s2[0] * (1.0f / HH) - mean * mean;
  h[(size_t)n * HH + c] += gelu_exact((y - mean) * rsqrtf(var + 1e-5f) * g2[c] + b2[c]);
}

// h * mask -> bf16 for the projection GEMM
__global__ __launch_bounds__(256) void k_hb(const float* __restrict__ h,
    const float* __restrict__ xmask, unsigned short* __restrict__ xb) {
  const int n = blockIdx.x, c = threadIdx.x;
  const size_t i = (size_t)n * HH + c;
  xb[i] = f2bf(h[i] * xmask[n]);
}

// 64(row-padded 58) x 256 projection GEMM, 4 waves/block, LDS-staged X + output
__global__ __launch_bounds__(128) void k_gemmp(const unsigned short* __restrict__ pwb,
    const unsigned short* __restrict__ xb, const float* __restrict__ pb,
    float* __restrict__ pout) {
  __shared__ unsigned short xs[32][HH];   // 16 KB
  __shared__ float tile[32][64];          // 8 KB
  const int tid = threadIdx.x, wave = tid >> 5, lane = tid & 31;
  const int lhi = lane >> 4, llo = lane & 15;
  const int o0 = wave * 16;
  const int n0 = blockIdx.x * 32;

#pragma unroll
  for (int j = 0; j < 8; ++j) {
    const int q  = tid + j * 128;
    const int rr = q >> 5;
    const int cc = (q & 31) * 8;
    g2l16(xb + (size_t)(n0 + rr) * HH + cc, &xs[rr][cc]);
  }
  wait_async0();
  __syncthreads();

  v8f c0, c1;
#pragma unroll
  for (int r = 0; r < 8; ++r) {
    const int o = o0 + r + 8 * lhi;
    const float bv = (o < 58) ? pb[o] : 0.0f;
    c0[r] = bv; c1[r] = bv;
  }
  const int row = o0 + llo;
#pragma unroll
  for (int kk = 0; kk < 8; ++kk) {
    const int koff = kk * 32 + lhi * 16;
    v16bf a  = *(const v16bf*)(pwb + row * HH + koff);
    v16bf b0 = *(const v16bf*)(&xs[llo][koff]);
    v16bf b1 = *(const v16bf*)(&xs[16 + llo][koff]);
    c0 = __builtin_amdgcn_wmma_f32_16x16x32_bf16(false, a, false, b0, (short)0, c0, false, false);
    c1 = __builtin_amdgcn_wmma_f32_16x16x32_bf16(false, a, false, b1, (short)0, c1, false, false);
  }
  float* d0 = &tile[llo][o0 + 8 * lhi];
  *(float4*)d0       = make_float4(c0[0], c0[1], c0[2], c0[3]);
  *(float4*)(d0 + 4) = make_float4(c0[4], c0[5], c0[6], c0[7]);
  float* d1 = &tile[16 + llo][o0 + 8 * lhi];
  *(float4*)d1       = make_float4(c1[0], c1[1], c1[2], c1[3]);
  *(float4*)(d1 + 4) = make_float4(c1[4], c1[5], c1[6], c1[7]);
  __syncthreads();
#pragma unroll
  for (int j = 0; j < 4; ++j) {
    const int q   = tid + j * 128;
    const int rr  = q >> 4;           // 16 chunks per 64-float row
    const int cc4 = (q & 15) * 4;
    l2g16(pout + (size_t)(n0 + rr) * 64 + cc4, &tile[rr][cc4]);
  }
}

__global__ void k_zero32(float* p) { p[threadIdx.x] = 0.0f; }

// out[:, :2] = x0 * mask
__global__ __launch_bounds__(256) void k_copyx0(const float* __restrict__ x,
    const float* __restrict__ xmask, float* __restrict__ out) {
  const int q = blockIdx.x * 256 + threadIdx.x;   // over B*2*T
  const int b = q >> 13, ch = (q >> 12) & 1, t = q & (TT - 1);
  const size_t idx = ((size_t)(b * 4 + ch)) * TT + t;
  out[idx] = x[idx] * xmask[b * TT + t];
}

// rational-quadratic spline on x1; bin walk fully in registers;
// per-block (single-b) LDS reduction of log|det| -> atomicAdd
__global__ __launch_bounds__(256) void k_spline(const float* __restrict__ x,
    const float* __restrict__ xmask, const float* __restrict__ pout,
    float* __restrict__ out, float* __restrict__ logdet) {
  __shared__ float sred[256];
  const int q = blockIdx.x * 256 + threadIdx.x;   // b*2*T + c*T + t
  const int b = q >> 13;
  const int c = (q >> 12) & 1;
  const int t = q & (TT - 1);
  const int n = b * TT + t;
  const float m = xmask[n];
  const float x1v = x[((size_t)(b * 4 + 2 + c)) * TT + t];
  const float* pp = pout + (size_t)n * 64 + c * 29;
  const float scale = 0.0625f;   // 1/sqrt(256)
  float uw[NBINS], uh[NBINS], ud[NBINS - 1];
#pragma unroll
  for (int j = 0; j < NBINS; ++j) uw[j] = pp[j] * scale * m;
#pragma unroll
  for (int j = 0; j < NBINS; ++j) uh[j] = pp[NBINS + j] * scale * m;
#pragma unroll
  for (int j = 0; j < NBINS - 1; ++j) ud[j] = pp[2 * NBINS + j] * m;
  float mw = uw[0], mh = uh[0];
#pragma unroll
  for (int j = 1; j < NBINS; ++j) { mw = fmaxf(mw, uw[j]); mh = fmaxf(mh, uh[j]); }
  float swn = 0.f, shn = 0.f;
#pragma unroll
  for (int j = 0; j < NBINS; ++j) {
    uw[j] = __expf(uw[j] - mw); swn += uw[j];
    uh[j] = __expf(uh[j] - mh); shn += uh[j];
  }
  const float rw = 1.0f / swn, rh = 1.0f / shn;
  const float xc = fminf(fmaxf(x1v, -TAILB), TAILB);
  float accW = 0.f, accH = 0.f, cwL = -TAILB, chL = -TAILB;
  float in_cw = -TAILB, in_w = 1.f, in_ch = -TAILB, in_h = 1.f, dk = 1.f, dk1 = 1.f;
  int found = 0;
#pragma unroll
  for (int j = 0; j < NBINS; ++j) {
    const float wj = 0.001f + 0.99f * uw[j] * rw;
    const float hj = 0.001f + 0.99f * uh[j] * rh;
    accW += wj; accH += hj;
    const float cwR = (j == NBINS - 1) ? TAILB : 2.0f * TAILB * accW - TAILB;
    const float chR = (j == NBINS - 1) ? TAILB : 2.0f * TAILB * accH - TAILB;
    const float dL  = (j == 0)         ? 1.0f : 0.001f + softplusf(ud[j - 1]);
    const float dR  = (j == NBINS - 1) ? 1.0f : 0.001f + softplusf(ud[j]);
    const int sel = (!found) && ((xc < cwR) || (j == NBINS - 1));
    if (sel) {
      found = 1;
      in_cw = cwL; in_w = cwR - cwL;
      in_ch = chL; in_h = chR - chL;
      dk = dL; dk1 = dR;
    }
    cwL = cwR; chL = chR;
  }
  const float theta = (xc - in_cw) / in_w;
  const float t1m = theta * (1.0f - theta);
  const float delta = in_h / in_w;
  const float num = in_h * (delta * theta * theta + dk * t1m);
  const float den = delta + (dk + dk1 - 2.0f * delta) * t1m;
  float y = in_ch + num / den;
  const float omt = 1.0f - theta;
  const float dnum = delta * delta * (dk1 * theta * theta + 2.0f * delta * t1m + dk * omt * omt);
  float lad = __logf(dnum) - 2.0f * __logf(den);
  const int inside = (x1v >= -TAILB) && (x1v <= TAILB);
  y = inside ? y : x1v;
  lad = inside ? lad : 0.0f;
  out[((size_t)(b * 4 + 2 + c)) * TT + t] = y * m;
  sred[threadIdx.x] = lad * m;
  __syncthreads();
  for (int s = 128; s > 0; s >>= 1) {
    if (threadIdx.x < s) sred[threadIdx.x] += sred[threadIdx.x + s];
    __syncthreads();
  }
  if (threadIdx.x == 0) atomicAdd(&logdet[b], sred[0]);
}

extern "C" void kernel_launch(void* const* d_in, const int* in_sizes, int n_in,
                              void* d_out, int out_size, void* d_ws, size_t ws_size,
                              hipStream_t stream) {
  const float* x      = (const float*)d_in[0];
  const float* xmask  = (const float*)d_in[1];
  const float* pre_w  = (const float*)d_in[2];
  const float* pre_b  = (const float*)d_in[3];
  const float* sep_w  = (const float*)d_in[4];
  const float* sep_b  = (const float*)d_in[5];
  const float* conv1w = (const float*)d_in[6];
  const float* conv1b = (const float*)d_in[7];
  const float* n1g    = (const float*)d_in[8];
  const float* n1b    = (const float*)d_in[9];
  const float* n2g    = (const float*)d_in[10];
  const float* n2b    = (const float*)d_in[11];
  const float* projw  = (const float*)d_in[12];
  const float* projb  = (const float*)d_in[13];
  (void)in_sizes; (void)n_in; (void)out_size; (void)ws_size;

  char* ws = (char*)d_ws;
  float*          h   = (float*)ws;                                        // N*256 f32
  unsigned short* xb  = (unsigned short*)(ws + (size_t)NTOT * HH * 4);     // N*256 bf16
  float*          yt  = (float*)(ws + (size_t)NTOT * HH * 6);              // N*256 f32 (also proj out)
  unsigned short* wb  = (unsigned short*)(ws + (size_t)NTOT * HH * 10);    // 256*256 bf16
  unsigned short* pwb = wb + HH * HH;                                      // 64*256 bf16
  float* out    = (float*)d_out;
  float* logdet = out + (size_t)BB * 4 * TT;

  k_pre<<<NTOT, 256, 0, stream>>>(x, pre_w, pre_b, h);
  int dil = 1;
  for (int i = 0; i < 4; ++i) {
    k_dwln<<<NTOT, 256, 0, stream>>>(h, xmask, sep_w + i * HH * 3, sep_b + i * HH,
                                     n1g + i * HH, n1b + i * HH, xb, dil);
    k_cvt<<<(HH * HH) / 256, 256, 0, stream>>>(conv1w + (size_t)i * HH * HH, wb, HH * HH);
    k_gemm<<<dim3(NTOT / 32, 2), 256, 0, stream>>>(wb, xb, conv1b + i * HH, yt);
    k_ln2<<<NTOT, 256, 0, stream>>>(yt, n2g + i * HH, n2b + i * HH, h);
    dil *= 3;
  }
  k_hb<<<NTOT, 256, 0, stream>>>(h, xmask, xb);
  k_projw<<<64, 256, 0, stream>>>(projw, pwb);
  k_gemmp<<<NTOT / 32, 128, 0, stream>>>(pwb, xb, projb, yt);
  k_zero32<<<1, 32, 0, stream>>>(logdet);
  k_copyx0<<<(BB * 2 * TT) / 256, 256, 0, stream>>>(x, xmask, out);
  k_spline<<<(BB * 2 * TT) / 256, 256, 0, stream>>>(x, xmask, yt, out, logdet);
}
